// VQVAE_32152125177958
// MI455X (gfx1250) — matrix operand order repaired and checked by reading, not compiled, not run on hardware
//
#include <hip/hip_runtime.h>

// ---------------------------------------------------------------------------
// VQVAE forward, fused for MI455X (gfx1250, wave32, WMMA f32_16x16x32_f16).
//
//   prep_transpose : f32 [K][N] weight -> f16 [N][K] (WMMA B layout)
//   prep_codebook  : E f32 -> Eh f16 (already [N][K]) + esq[j] = 0.5*|E_j|^2
//   enc_vq_kernel  : x -> h(LDS) -> z/flat(LDS) -> VQ argmin -> idx + loss partial
//   dec_kernel     : idx -> q (gathered from Eh) -> d(LDS) -> x_hat (global)
//   loss_final     : sum partials * 1.25 / (B*LATENT) -> d_out tail
//
// MTILE=128 rows / 1024 threads (32 waves): each weight B-fragment is loaded
// exactly once per workgroup (wave owns one N-block, iterates all 8 M-blocks),
// so L2 weight traffic = weight_bytes * BATCH/128 ~ 1.45 GB, at parity with
// the 512 MB HBM floor. All intermediates (h, z, d) live in LDS (~225 KB enc,
// ~130 KB dec; gfx1250 allows 320 KB/workgroup).
// ---------------------------------------------------------------------------

typedef _Float16 v8h  __attribute__((ext_vector_type(8)));
typedef _Float16 v16h __attribute__((ext_vector_type(16)));
typedef float    v8f  __attribute__((ext_vector_type(8)));

#define BATCH   65536
#define IN_DIM  1024
#define H_ENC   512
#define LATENT  256
#define KDIM    512
#define H_DEC   512
#define MTILE   128       // batch rows per workgroup
#define NTHR    1024      // 32 waves
#define NWG     (BATCH / MTILE)
#define FTILE   (MTILE / 2)   // flat rows per workgroup (64)
#define LDX     72        // x staging row stride (64 + 8 pad)
#define LDH     (H_ENC + 8)
#define LDF     (KDIM + 8)
#define LDD     (H_DEC + 8)

// ---- WMMA helpers ----------------------------------------------------------

__device__ __forceinline__ v8f wmma_f16(v16h a, v16h b, v8f c) {
  return __builtin_amdgcn_wmma_f32_16x16x32_f16(
      false, a, false, b, (short)0, c, false, false);
}

// A-fragment: caller passes p = rowbase + k0 + (lane>>4)*16
__device__ __forceinline__ v16h frag_a(const _Float16* p) {
  v8h lo = *(const v8h*)(p);
  v8h hi = *(const v8h*)(p + 8);
  return __builtin_shufflevector(lo, hi, 0,1,2,3,4,5,6,7,8,9,10,11,12,13,14,15);
}

// B-fragment: caller passes p = bt + n*ldk + k0 + (lane>>4)*16
__device__ __forceinline__ v16h frag_b(const _Float16* p) {
  v8h lo = *(const v8h*)(p);
  v8h hi = *(const v8h*)(p + 8);
  return __builtin_shufflevector(lo, hi, 0,1,2,3,4,5,6,7,8,9,10,11,12,13,14,15);
}

// ---- weight prep -----------------------------------------------------------

__global__ __launch_bounds__(256) void prep_transpose(
    const float* __restrict__ src, _Float16* __restrict__ dst, int K, int N) {
  int n = blockIdx.x;
  for (int k = threadIdx.x; k < K; k += 256)
    dst[(size_t)n * K + k] = (_Float16)src[(size_t)k * N + n];
}

__global__ __launch_bounds__(256) void prep_codebook(
    const float* __restrict__ E, _Float16* __restrict__ Eh, float* __restrict__ esq) {
  __shared__ float red[256];
  int j = blockIdx.x, t = threadIdx.x;
  float s = 0.f;
  for (int k = t; k < KDIM; k += 256) {
    float v = E[(size_t)j * KDIM + k];
    Eh[(size_t)j * KDIM + k] = (_Float16)v;
    s += v * v;
  }
  red[t] = s; __syncthreads();
  for (int off = 128; off > 0; off >>= 1) {
    if (t < off) red[t] += red[t + off];
    __syncthreads();
  }
  if (t == 0) esq[j] = 0.5f * red[0];
}

// ---- fused encoder + vector quantizer --------------------------------------

__global__ __launch_bounds__(NTHR) void enc_vq_kernel(
    const float* __restrict__ x,
    const _Float16* __restrict__ W1h, const float* __restrict__ b1,
    const _Float16* __restrict__ W2h, const float* __restrict__ b2,
    const _Float16* __restrict__ Eh,  const float* __restrict__ esq,
    const float* __restrict__ E32,
    int* __restrict__ idx_out, float* __restrict__ loss_partial) {
  const int tid = threadIdx.x, lane = tid & 31, wv = tid >> 5;   // wv 0..31
  const int jloc = lane & 15, hi = lane >> 4;
  const int r0 = blockIdx.x * MTILE;

  __shared__ __attribute__((aligned(16))) _Float16 xs[MTILE][LDX];   // x chunk (K=64)
  __shared__ __attribute__((aligned(16))) _Float16 hs[MTILE][LDH];   // h tile f16
  __shared__ __attribute__((aligned(16))) _Float16 fs[FTILE][LDF];   // flat(=z) tile f16
  __shared__ float sc[FTILE][16];
  __shared__ int   sj[FTILE][16];
  __shared__ int   bj[FTILE];
  __shared__ float red[NTHR];

  const v8f vzero = {0.f,0.f,0.f,0.f,0.f,0.f,0.f,0.f};

  // ===== GEMM1: h = relu(x @ W1 + b1), out 128x512, K=1024 ===================
  // wave wv owns N-block wv; iterates all 8 M-blocks -> each W1 tile loaded
  // exactly once per workgroup.
  {
    v8f acc[8];
    #pragma unroll
    for (int i = 0; i < 8; ++i) acc[i] = vzero;
    const int n = wv * 16 + jloc;

    for (int kk = 0; kk < IN_DIM; kk += 64) {
      __syncthreads();                       // xs reads of prev chunk done
      {   // stage x chunk [128 rows x 64 k] -> f16; 8 floats per thread
        int row = tid >> 3, col = (tid & 7) * 8;
        const float* xp = x + (size_t)(r0 + row) * IN_DIM + kk + col;
        const float4 v0 = *(const float4*)(xp);
        const float4 v1 = *(const float4*)(xp + 4);
        xs[row][col + 0] = (_Float16)v0.x;  xs[row][col + 1] = (_Float16)v0.y;
        xs[row][col + 2] = (_Float16)v0.z;  xs[row][col + 3] = (_Float16)v0.w;
        xs[row][col + 4] = (_Float16)v1.x;  xs[row][col + 5] = (_Float16)v1.y;
        xs[row][col + 6] = (_Float16)v1.z;  xs[row][col + 7] = (_Float16)v1.w;
      }
      __syncthreads();
      #pragma unroll
      for (int k2 = 0; k2 < 64; k2 += 32) {
        v16h b = frag_b(W1h + (size_t)n * IN_DIM + kk + k2 + hi * 16);
        #pragma unroll
        for (int mb = 0; mb < 8; ++mb) {
          v16h a = frag_a(&xs[mb * 16 + jloc][k2 + hi * 16]);
          acc[mb] = wmma_f16(a, b, acc[mb]);
        }
      }
    }
    __syncthreads();
    float bias = b1[n];
    #pragma unroll
    for (int mb = 0; mb < 8; ++mb)
      #pragma unroll
      for (int v = 0; v < 8; ++v) {
        int m = mb * 16 + v + 8 * hi;
        float val = acc[mb][v] + bias;
        hs[m][n] = (_Float16)(val > 0.f ? val : 0.f);
      }
  }
  __syncthreads();

  // ===== GEMM2: z = relu(h @ W2 + b2), out 128x256, K=512 ====================
  // wave wv: N-block wv&15, M-blocks (wv>>4)*4 + {0..3}; written as flat [64][512]
  {
    v8f acc[4];
    #pragma unroll
    for (int i = 0; i < 4; ++i) acc[i] = vzero;
    const int n = (wv & 15) * 16 + jloc;
    const int mbg = (wv >> 4) * 4;
    for (int kk = 0; kk < H_ENC; kk += 32) {
      v16h b = frag_b(W2h + (size_t)n * H_ENC + kk + hi * 16);
      #pragma unroll
      for (int i = 0; i < 4; ++i) {
        v16h a = frag_a(&hs[(mbg + i) * 16 + jloc][kk + hi * 16]);
        acc[i] = wmma_f16(a, b, acc[i]);
      }
    }
    __syncthreads();
    float bias = b2[n];
    #pragma unroll
    for (int i = 0; i < 4; ++i)
      #pragma unroll
      for (int v = 0; v < 8; ++v) {
        int r = (mbg + i) * 16 + v + 8 * hi;        // z row in tile (0..127)
        float val = acc[i][v] + bias;
        fs[r >> 1][(r & 1) * LATENT + n] = (_Float16)(val > 0.f ? val : 0.f);
      }
  }
  __syncthreads();

  // ===== VQ: scores = flat @ Eh^T - 0.5|E_j|^2, argmax over 256 codewords ====
  // out 64x256: wave wv: N-block wv&15, flat M-blocks (wv>>4)*2 + {0,1}
  {
    v8f acc[2];
    acc[0] = vzero; acc[1] = vzero;
    const int jj = (wv & 15) * 16 + jloc;
    const int fbg = (wv >> 4) * 2;
    for (int kk = 0; kk < KDIM; kk += 32) {
      v16h b = frag_b(Eh + (size_t)jj * KDIM + kk + hi * 16);
      #pragma unroll
      for (int i = 0; i < 2; ++i) {
        v16h a = frag_a(&fs[(fbg + i) * 16 + jloc][kk + hi * 16]);
        acc[i] = wmma_f16(a, b, acc[i]);
      }
    }
    float e = esq[jj];
    #pragma unroll
    for (int i = 0; i < 2; ++i)
      #pragma unroll
      for (int v = 0; v < 8; ++v) {
        float bs = acc[i][v] - e; int bi = jj;
        // reduce across the 16 lanes of each half-wave (bit4 untouched)
        #pragma unroll
        for (int m = 1; m < 16; m <<= 1) {
          float os = __shfl_xor(bs, m, 32);
          int   oi = __shfl_xor(bi, m, 32);
          if (os > bs || (os == bs && oi < bi)) { bs = os; bi = oi; }
        }
        if (jloc == 0) {
          int fm = (fbg + i) * 16 + v + 8 * hi;
          sc[fm][wv & 15] = bs; sj[fm][wv & 15] = bi;
        }
      }
  }
  __syncthreads();
  if (tid < FTILE) {
    float bs = sc[tid][0]; int bi = sj[tid][0];
    #pragma unroll
    for (int w = 1; w < 16; ++w) {
      float s = sc[tid][w]; int i = sj[tid][w];
      if (s > bs || (s == bs && i < bi)) { bs = s; bi = i; }
    }
    bj[tid] = bi;
    idx_out[blockIdx.x * FTILE + tid] = bi;
  }
  __syncthreads();

  // ===== loss partial: sum (E[bj[fm]][k] - flat[fm][k])^2 ====================
  {
    int fm = tid & (FTILE - 1), ks = (tid >> 6) * 32;   // 16 threads x 32 k per fm
    const float* er = E32 + (size_t)bj[fm] * KDIM + ks;
    float s = 0.f;
    #pragma unroll 8
    for (int k = 0; k < 32; ++k) {
      float d = er[k] - (float)fs[fm][ks + k];
      s += d * d;
    }
    red[tid] = s; __syncthreads();
    for (int off = NTHR / 2; off > 0; off >>= 1) {
      if (tid < off) red[tid] += red[tid + off];
      __syncthreads();
    }
    if (tid == 0) loss_partial[blockIdx.x] = red[0];
  }
}

// ---- fused decoder ---------------------------------------------------------

__global__ __launch_bounds__(NTHR) void dec_kernel(
    const int* __restrict__ idx, const _Float16* __restrict__ Eh,
    const _Float16* __restrict__ Wd1h, const float* __restrict__ bd1,
    const _Float16* __restrict__ Wd2h, const float* __restrict__ bd2,
    float* __restrict__ out) {
  const int tid = threadIdx.x, lane = tid & 31, wv = tid >> 5;
  const int jloc = lane & 15, hi = lane >> 4;
  const int r0 = blockIdx.x * MTILE, f0 = r0 >> 1;

  __shared__ __attribute__((aligned(16))) _Float16 dsh[MTILE][LDD];
  __shared__ int bjs[FTILE];

  if (tid < FTILE) bjs[tid] = idx[f0 + tid];
  __syncthreads();

  const v8f vzero = {0.f,0.f,0.f,0.f,0.f,0.f,0.f,0.f};

  // ===== GEMM: d = relu(q @ Wd1 + bd1), out 128x512, K=256 ===================
  // q row r = E[idx[r/2]][(r%2)*256 ...] gathered straight from Eh (f16);
  // wave wv owns N-block wv, iterates all 8 M-blocks.
  {
    const _Float16* qrow[8];
    #pragma unroll
    for (int mb = 0; mb < 8; ++mb) {
      int r = mb * 16 + jloc;
      qrow[mb] = Eh + (size_t)bjs[r >> 1] * KDIM + (r & 1) * LATENT;
    }
    v8f acc[8];
    #pragma unroll
    for (int i = 0; i < 8; ++i) acc[i] = vzero;
    const int n = wv * 16 + jloc;
    for (int kk = 0; kk < LATENT; kk += 32) {
      v16h b = frag_b(Wd1h + (size_t)n * LATENT + kk + hi * 16);
      #pragma unroll
      for (int mb = 0; mb < 8; ++mb) {
        v16h a = frag_a(qrow[mb] + kk + hi * 16);
        acc[mb] = wmma_f16(a, b, acc[mb]);
      }
    }
    float bias = bd1[n];
    #pragma unroll
    for (int mb = 0; mb < 8; ++mb)
      #pragma unroll
      for (int v = 0; v < 8; ++v) {
        int m = mb * 16 + v + 8 * hi;
        float val = acc[mb][v] + bias;
        dsh[m][n] = (_Float16)(val > 0.f ? val : 0.f);
      }
  }
  __syncthreads();

  // ===== GEMM: x_hat = d @ Wd2 + bd2, out 128x1024, K=512, two N-passes ======
  for (int p = 0; p < 2; ++p) {
    v8f acc[8];
    #pragma unroll
    for (int i = 0; i < 8; ++i) acc[i] = vzero;
    const int n = (p * 32 + wv) * 16 + jloc;
    for (int kk = 0; kk < H_DEC; kk += 32) {
      v16h b = frag_b(Wd2h + (size_t)n * H_DEC + kk + hi * 16);
      #pragma unroll
      for (int mb = 0; mb < 8; ++mb) {
        v16h a = frag_a(&dsh[mb * 16 + jloc][kk + hi * 16]);
        acc[mb] = wmma_f16(a, b, acc[mb]);
      }
    }
    float bias = bd2[n];
    #pragma unroll
    for (int mb = 0; mb < 8; ++mb)
      #pragma unroll
      for (int v = 0; v < 8; ++v) {
        int row = r0 + mb * 16 + v + 8 * hi;
        out[(size_t)row * IN_DIM + n] = acc[mb][v] + bias;
      }
  }
}

// ---- loss finalize ---------------------------------------------------------

__global__ __launch_bounds__(256) void loss_final(
    const float* __restrict__ partial, int n, float* __restrict__ out_loss) {
  __shared__ float red[256];
  int t = threadIdx.x;
  float s = 0.f;
  for (int i = t; i < n; i += 256) s += partial[i];
  red[t] = s; __syncthreads();
  for (int off = 128; off > 0; off >>= 1) {
    if (t < off) red[t] += red[t + off];
    __syncthreads();
  }
  if (t == 0) out_loss[0] = red[0] * 1.25f / ((float)BATCH * (float)LATENT);
}

// ---------------------------------------------------------------------------

extern "C" void kernel_launch(void* const* d_in, const int* in_sizes, int n_in,
                              void* d_out, int out_size, void* d_ws, size_t ws_size,
                              hipStream_t stream) {
  const float* x   = (const float*)d_in[0];
  const float* W1  = (const float*)d_in[1];
  const float* b1  = (const float*)d_in[2];
  const float* W2  = (const float*)d_in[3];
  const float* b2  = (const float*)d_in[4];
  const float* E   = (const float*)d_in[5];
  const float* Wd1 = (const float*)d_in[6];
  const float* bd1 = (const float*)d_in[7];
  const float* Wd2 = (const float*)d_in[8];
  const float* bd2 = (const float*)d_in[9];

  float* xhat = (float*)d_out;
  float* loss = xhat + (size_t)BATCH * IN_DIM;

  char* w = (char*)d_ws;
  auto alloc = [&](size_t bytes) -> void* {
    void* p = (void*)w;
    w += (bytes + 255) & ~(size_t)255;
    return p;
  };
  _Float16* W1h  = (_Float16*)alloc((size_t)H_ENC  * IN_DIM * 2);  // [512][1024]
  _Float16* W2h  = (_Float16*)alloc((size_t)LATENT * H_ENC  * 2);  // [256][512]
  _Float16* Eh   = (_Float16*)alloc((size_t)LATENT * KDIM   * 2);  // [256][512]
  _Float16* Wd1h = (_Float16*)alloc((size_t)H_DEC  * LATENT * 2);  // [512][256]
  _Float16* Wd2h = (_Float16*)alloc((size_t)IN_DIM * H_DEC  * 2);  // [1024][512]
  float* esq     = (float*)alloc(LATENT * 4);
  int*   idx     = (int*)alloc((size_t)(BATCH / 2) * 4);
  float* lpart   = (float*)alloc((size_t)NWG * 4);

  prep_transpose<<<H_ENC,  256, 0, stream>>>(W1,  W1h,  IN_DIM, H_ENC);
  prep_transpose<<<LATENT, 256, 0, stream>>>(W2,  W2h,  H_ENC,  LATENT);
  prep_transpose<<<H_DEC,  256, 0, stream>>>(Wd1, Wd1h, LATENT, H_DEC);
  prep_transpose<<<IN_DIM, 256, 0, stream>>>(Wd2, Wd2h, H_DEC,  IN_DIM);
  prep_codebook<<<LATENT,  256, 0, stream>>>(E, Eh, esq);

  enc_vq_kernel<<<NWG, NTHR, 0, stream>>>(x, W1h, b1, W2h, b2, Eh, esq, E, idx, lpart);
  dec_kernel<<<NWG, NTHR, 0, stream>>>(idx, Eh, Wd1h, bd1, Wd2h, bd2, xhat);
  loss_final<<<1, 256, 0, stream>>>(lpart, NWG, loss);
}
